// Linear2Bit_13331578486911
// MI455X (gfx1250) — matrix operand order repaired and checked by reading, not compiled, not run on hardware
//
#include <hip/hip_runtime.h>

#define OUT_DIM 11008
#define IN_DIM  4096
#define BATCH   32

typedef __attribute__((ext_vector_type(16))) _Float16 v16h;
typedef __attribute__((ext_vector_type(8)))  float    v8f;

// ---------------------------------------------------------------------------
// Kernel 1: split fp32 activations into hi/lo f16 parts (x = hi + lo).
// Ternary weights are exact in f16, so the only fp32->f16 loss is in x;
// the hi/lo split recovers ~fp32 accuracy with two WMMAs per weight tile.
// ---------------------------------------------------------------------------
__global__ void cvt_x_kernel(const float* __restrict__ x,
                             _Float16* __restrict__ xhi,
                             _Float16* __restrict__ xlo, int n) {
  int i = blockIdx.x * blockDim.x + threadIdx.x;
  if (i < n) {
    float v = x[i];
    _Float16 h = (_Float16)v;
    xhi[i] = h;
    xlo[i] = (_Float16)(v - (float)h);
  }
}

// ---------------------------------------------------------------------------
// Kernel 2: ternary-weight GEMM via v_wmma_f32_16x16x32_f16.
//   block  = 256 threads = 8 waves (wave32)
//   block tile: 16 OUT rows x 32 batch cols, K split 8 ways across waves
//   per wave: 16 K-steps of 32; A = weights (int32 ternary -> f16),
//             B = x_hi / x_lo (f16, preconverted), C = f32 accum
// ---------------------------------------------------------------------------
__launch_bounds__(256)
__global__ void ternary_gemm_kernel(const int*      __restrict__ wq,
                                    const float*    __restrict__ scale,
                                    const float*    __restrict__ bias,
                                    const _Float16* __restrict__ xhi,
                                    const _Float16* __restrict__ xlo,
                                    float*          __restrict__ out) {
  __shared__ float red[8 * 2 * 32 * 8];   // [wave][acc][lane][vgpr] = 16 KB

  const int lane = threadIdx.x & 31;
  const int wv   = threadIdx.x >> 5;          // wave id 0..7
  const int m    = lane & 15;                 // A row within tile
  const int h    = lane >> 4;                 // lane half (K grouping)
  const int out_base = blockIdx.x * 16;

  const int kslice = IN_DIM / 8;              // 512
  const int k0     = wv * kslice;

  const int*      wrow = wq  + (size_t)(out_base + m) * IN_DIM;
  const _Float16* bh0p = xhi + (size_t)(lane & 15) * IN_DIM;        // batch 0..15
  const _Float16* bl0p = xlo + (size_t)(lane & 15) * IN_DIM;
  const _Float16* bh1p = bh0p + (size_t)16 * IN_DIM;                // batch 16..31
  const _Float16* bl1p = bl0p + (size_t)16 * IN_DIM;

  v8f c0 = {};   // N = batch 0..15
  v8f c1 = {};   // N = batch 16..31

  for (int kk = k0; kk < k0 + kslice; kk += 32) {
    // ---- A tile: 16x32 ternary int32 weights -> v16h (f16 exact) ----
    // lane half h=0 owns K = kk+{0..15}; h=1 owns K = kk+{16..31}
    const int b1 = kk + 16 * h;
    const int b2 = kk + 16 * h + 8;
    int4 w0 = *(const int4*)(wrow + b1);
    int4 w1 = *(const int4*)(wrow + b1 + 4);
    int4 w2 = *(const int4*)(wrow + b2);
    int4 w3 = *(const int4*)(wrow + b2 + 4);
    v16h a;
    a[0]  = (_Float16)(float)w0.x;  a[1]  = (_Float16)(float)w0.y;
    a[2]  = (_Float16)(float)w0.z;  a[3]  = (_Float16)(float)w0.w;
    a[4]  = (_Float16)(float)w1.x;  a[5]  = (_Float16)(float)w1.y;
    a[6]  = (_Float16)(float)w1.z;  a[7]  = (_Float16)(float)w1.w;
    a[8]  = (_Float16)(float)w2.x;  a[9]  = (_Float16)(float)w2.y;
    a[10] = (_Float16)(float)w2.z;  a[11] = (_Float16)(float)w2.w;
    a[12] = (_Float16)(float)w3.x;  a[13] = (_Float16)(float)w3.y;
    a[14] = (_Float16)(float)w3.z;  a[15] = (_Float16)(float)w3.w;

    // ---- B tiles: 32x16 f16, lane owns column n=lane&15, K = kk+16h+{0..15}
    const int bofs = kk + 16 * h;
    v16h bh_0 = *(const v16h*)(bh0p + bofs);
    v16h bl_0 = *(const v16h*)(bl0p + bofs);
    v16h bh_1 = *(const v16h*)(bh1p + bofs);
    v16h bl_1 = *(const v16h*)(bl1p + bofs);

    // ---- MMA: hi + lo chained into the same f32 accumulator ----
    c0 = __builtin_amdgcn_wmma_f32_16x16x32_f16(false, a, false, bh_0,
                                                (short)0, c0, false, false);
    c0 = __builtin_amdgcn_wmma_f32_16x16x32_f16(false, a, false, bl_0,
                                                (short)0, c0, false, false);
    c1 = __builtin_amdgcn_wmma_f32_16x16x32_f16(false, a, false, bh_1,
                                                (short)0, c1, false, false);
    c1 = __builtin_amdgcn_wmma_f32_16x16x32_f16(false, a, false, bl_1,
                                                (short)0, c1, false, false);
  }

  // ---- cross-wave K reduction through LDS ----
  {
    float* p0 = &red[(size_t)((wv * 2 + 0) * 32 + lane) * 8];
    float* p1 = &red[(size_t)((wv * 2 + 1) * 32 + lane) * 8];
#pragma unroll
    for (int r = 0; r < 8; ++r) { p0[r] = c0[r]; p1[r] = c1[r]; }
  }
  __syncthreads();

  if (wv < 2) {                       // wave 0 -> batches 0..15, wave 1 -> 16..31
    float acc[8] = {0.f, 0.f, 0.f, 0.f, 0.f, 0.f, 0.f, 0.f};
#pragma unroll
    for (int ww = 0; ww < 8; ++ww) {
      const float* p = &red[(size_t)((ww * 2 + wv) * 32 + lane) * 8];
#pragma unroll
      for (int r = 0; r < 8; ++r) acc[r] += p[r];
    }
    // C/D layout: lane l, vgpr r -> (M = r + 8*(l>>4), N = l&15)
    const int n  = (lane & 15) + 16 * wv;   // batch index
    const int o0 = out_base + 8 * h;        // OUT row base for this lane
#pragma unroll
    for (int r = 0; r < 8; ++r) {
      out[(size_t)n * OUT_DIM + o0 + r] = acc[r] * scale[o0 + r] + bias[o0 + r];
    }
  }
}

// ---------------------------------------------------------------------------
extern "C" void kernel_launch(void* const* d_in, const int* in_sizes, int n_in,
                              void* d_out, int out_size, void* d_ws, size_t ws_size,
                              hipStream_t stream) {
  const float* x  = (const float*)d_in[0];   // [32, 4096] fp32
  const int*   wq = (const int*)  d_in[1];   // [11008, 4096] int32 ternary
  const float* sc = (const float*)d_in[2];   // [11008, 1] fp32
  const float* bs = (const float*)d_in[3];   // [11008] fp32
  float* out = (float*)d_out;                // [32, 11008] fp32

  // workspace: x_hi and x_lo f16 planes (2 * 32*4096*2 B = 512 KB)
  _Float16* xhi = (_Float16*)d_ws;
  _Float16* xlo = xhi + (size_t)BATCH * IN_DIM;

  const int n = BATCH * IN_DIM;
  cvt_x_kernel<<<(n + 255) / 256, 256, 0, stream>>>(x, xhi, xlo, n);

  ternary_gemm_kernel<<<OUT_DIM / 16, 256, 0, stream>>>(wq, sc, bs, xhi, xlo, out);
}
